// LSTMModel_20933670600966
// MI455X (gfx1250) — matrix operand order repaired and checked by reading, not compiled, not run on hardware
//
#include <hip/hip_runtime.h>

// ---------------- problem constants ----------------
#define IN_   64
#define Hh    256
#define FOURH 1024      // 4*H
#define TWOH  512
#define Bsz   64
#define Tsz   512
#define Lnum  5
#define OUTsz 48
#define MTOT  (Bsz * Tsz)   // 32768
#define LDSTRIDE 264        // 256 + 8 pad (bf16 elems): 528B rows, conflict-free b128 LDS reads

typedef __attribute__((ext_vector_type(16))) __bf16 v16bf;
typedef __attribute__((ext_vector_type(8)))  __bf16 v8bf;
typedef __attribute__((ext_vector_type(8)))  float  v8f;

// A-matrix fragment (16x32, ISA 7.12.2 dense-A table): lane half h holds
// K-chunks [k0+8h, +8) and [k0+16+8h, +8)  -> two 16B loads.
static __device__ __forceinline__ v16bf load_fragA(const __bf16* __restrict__ row,
                                                   int k0, int half) {
  v8bf lo = *(const v8bf*)(row + k0 + 8 * half);
  v8bf hi = *(const v8bf*)(row + k0 + 16 + 8 * half);
  return __builtin_shufflevector(lo, hi, 0, 1, 2, 3, 4, 5, 6, 7,
                                 8, 9, 10, 11, 12, 13, 14, 15);
}

// B-matrix fragment (32x16): lane half h holds
// K-chunks [k0+8h, +8) and [k0+16+8h, +8) (same order as A)  -> two 16B loads.
static __device__ __forceinline__ v16bf load_fragB(const __bf16* __restrict__ row,
                                                   int k0, int half) {
  v8bf lo = *(const v8bf*)(row + k0 + 8 * half);
  v8bf hi = *(const v8bf*)(row + k0 + 16 + 8 * half);
  return __builtin_shufflevector(lo, hi, 0, 1, 2, 3, 4, 5, 6, 7,
                                 8, 9, 10, 11, 12, 13, 14, 15);
}

static __device__ __forceinline__ float sigf(float x) {
  return 1.0f / (1.0f + __expf(-x));
}

// ---------------- fp32 -> bf16 convert ----------------
__global__ void k_f2bf(const float* __restrict__ src, __bf16* __restrict__ dst, int n) {
  for (int i = blockIdx.x * blockDim.x + threadIdx.x; i < n; i += gridDim.x * blockDim.x)
    dst[i] = (__bf16)src[i];
}

// bias_sum = b_ih + b_hh
__global__ void k_biassum(const float* __restrict__ a, const float* __restrict__ b,
                          float* __restrict__ dst, int n) {
  for (int i = blockIdx.x * blockDim.x + threadIdx.x; i < n; i += gridDim.x * blockDim.x)
    dst[i] = a[i] + b[i];
}

// ---------------- input projection GEMM (WMMA bf16) ----------------
// xg[dir][row][n] = sum_k A[row][k] * W[dir][n][k] + bias[dir][n]
// grid: (MTOT/32, 4, 2dirs), block: 128 (4 waves). Each wave: 32 rows x 64 cols
// (two 16-row tiles share every B fragment).
__global__ __launch_bounds__(128) void k_proj(const __bf16* __restrict__ A, int K,
                                              const __bf16* __restrict__ W,
                                              const float* __restrict__ bias,
                                              float* __restrict__ xg) {
  const int dir = blockIdx.z;
  const __bf16* Wd = W + (size_t)dir * FOURH * K;
  const float*  bd = bias + dir * FOURH;
  float*       xgd = xg + (size_t)dir * MTOT * FOURH;

  const int lane = threadIdx.x & 31;
  const int wave = threadIdx.x >> 5;
  const int mt   = blockIdx.x;                 // 32-row tile
  const int nb   = (blockIdx.y * 4 + wave) * 64;
  const int lrow = lane & 15;
  const int half = lane >> 4;

  const __bf16* Arow0 = A + (size_t)(mt * 32 + lrow) * K;
  const __bf16* Arow1 = A + (size_t)(mt * 32 + 16 + lrow) * K;

  v8f acc[2][4];
#pragma unroll
  for (int nt = 0; nt < 4; ++nt) {
    const float bv = bd[nb + nt * 16 + lrow];   // col value, same for all rows
#pragma unroll
    for (int j = 0; j < 8; ++j) { acc[0][nt][j] = bv; acc[1][nt][j] = bv; }
  }

  for (int k0 = 0; k0 < K; k0 += 32) {
    const v16bf a0 = load_fragA(Arow0, k0, half);
    const v16bf a1 = load_fragA(Arow1, k0, half);
#pragma unroll
    for (int nt = 0; nt < 4; ++nt) {
      const v16bf b = load_fragB(Wd + (size_t)(nb + nt * 16 + lrow) * K, k0, half);
      acc[0][nt] = __builtin_amdgcn_wmma_f32_16x16x32_bf16(false, a0, false, b,
                                                           (short)0, acc[0][nt], false, false);
      acc[1][nt] = __builtin_amdgcn_wmma_f32_16x16x32_bf16(false, a1, false, b,
                                                           (short)0, acc[1][nt], false, false);
    }
  }

#pragma unroll
  for (int m = 0; m < 2; ++m)
#pragma unroll
    for (int nt = 0; nt < 4; ++nt)
#pragma unroll
      for (int j = 0; j < 8; ++j) {
        const int row = mt * 32 + m * 16 + j + 8 * half;
        xgd[(size_t)row * FOURH + nb + nt * 16 + lrow] = acc[m][nt][j];
      }
}

// ---------------- recurrent LSTM scan (persistent WG per direction) ----------------
// grid = 2 (dir), block = 1024 (32 waves). Per step: gates = xg_t + h @ Whh^T,
// h kept in LDS (bf16), c in registers. Wave w: col-group cg=w&15 covers
// hidx in [16cg,16cg+16) for ALL 4 gates; m-pair mp=w>>4 covers b-tiles {2mp,2mp+1}.
__global__ __launch_bounds__(1024) void k_scan(const float* __restrict__ xg,
                                               const __bf16* __restrict__ Whh,
                                               float* __restrict__ hout_f32,
                                               __bf16* __restrict__ hout_bf,
                                               int write_f32, int write_bf) {
  __shared__ __align__(16) __bf16 hbuf[64 * LDSTRIDE];

  const int dir = blockIdx.x;
  const float*  xgd = xg + (size_t)dir * MTOT * FOURH;
  const __bf16* Wd  = Whh + (size_t)dir * FOURH * Hh;

  const int lane = threadIdx.x & 31;
  const int wave = threadIdx.x >> 5;
  const int cg   = wave & 15;
  const int mp   = wave >> 4;
  const int lrow = lane & 15;
  const int half = lane >> 4;
  const int hidx = cg * 16 + lrow;

  for (int i = threadIdx.x; i < 64 * LDSTRIDE; i += 1024) hbuf[i] = (__bf16)0.0f;

  float cst[2][8];
#pragma unroll
  for (int m = 0; m < 2; ++m)
#pragma unroll
    for (int j = 0; j < 8; ++j) cst[m][j] = 0.0f;

  __syncthreads();

  for (int step = 0; step < Tsz; ++step) {
    const int t = dir ? (Tsz - 1 - step) : step;

    // seed accumulators with xg_t (holds x·Wih + b_ih + b_hh)
    v8f acc[2][4];
#pragma unroll
    for (int m = 0; m < 2; ++m) {
      const int btile = (2 * mp + m) * 16;
#pragma unroll
      for (int g = 0; g < 4; ++g)
#pragma unroll
        for (int j = 0; j < 8; ++j) {
          const int b = btile + j + 8 * half;
          acc[m][g][j] = xgd[((size_t)b * Tsz + t) * FOURH + g * Hh + hidx];
        }
    }

    // cooperatively prefetch next step's xg rows (64 rows x 4KB = 2048 lines)
    if (step + 1 < Tsz) {
      const int tn = dir ? (Tsz - 2 - step) : (step + 1);
#pragma unroll
      for (int q = 0; q < 2; ++q) {
        const int li = threadIdx.x + q * 1024;   // 0..2047
        const int br = li >> 5;                  // row b
        const int lo = li & 31;                  // 128B line within row
        __builtin_prefetch((const char*)(xgd + ((size_t)br * Tsz + tn) * FOURH) + lo * 128,
                           0, 3);
      }
    }

    // gates += h_{t-1} @ Whh^T   (K = 256, 8 WMMA k-steps)
    for (int k0 = 0; k0 < Hh; k0 += 32) {
      const v16bf a0 = load_fragA(hbuf + (size_t)((2 * mp + 0) * 16 + lrow) * LDSTRIDE, k0, half);
      const v16bf a1 = load_fragA(hbuf + (size_t)((2 * mp + 1) * 16 + lrow) * LDSTRIDE, k0, half);
#pragma unroll
      for (int g = 0; g < 4; ++g) {
        const v16bf b = load_fragB(Wd + (size_t)(g * Hh + hidx) * Hh, k0, half);
        acc[0][g] = __builtin_amdgcn_wmma_f32_16x16x32_bf16(false, a0, false, b,
                                                            (short)0, acc[0][g], false, false);
        acc[1][g] = __builtin_amdgcn_wmma_f32_16x16x32_bf16(false, a1, false, b,
                                                            (short)0, acc[1][g], false, false);
      }
    }

    __syncthreads();   // all LDS reads of h_{t-1} complete before overwrite

    // LSTM cell (gate order i,f,g,o) + write h
#pragma unroll
    for (int m = 0; m < 2; ++m) {
      const int btile = (2 * mp + m) * 16;
#pragma unroll
      for (int j = 0; j < 8; ++j) {
        const int b = btile + j + 8 * half;
        const float iv = sigf(acc[m][0][j]);
        const float fv = sigf(acc[m][1][j]);
        const float gv = tanhf(acc[m][2][j]);
        const float ov = sigf(acc[m][3][j]);
        const float c  = fv * cst[m][j] + iv * gv;
        cst[m][j] = c;
        const float h = ov * tanhf(c);
        hbuf[b * LDSTRIDE + hidx] = (__bf16)h;
        const size_t oidx = ((size_t)b * Tsz + t) * TWOH + dir * Hh + hidx;
        if (write_f32) hout_f32[oidx] = h;
        if (write_bf)  hout_bf[oidx]  = (__bf16)h;
      }
    }
    __syncthreads();
  }
}

// ---------------- LN + FC1(relu) + FC2(sigmoid) head ----------------
__global__ __launch_bounds__(256) void k_head(const float* __restrict__ hfin,
                                              const float* __restrict__ lng,
                                              const float* __restrict__ lnb,
                                              const float* __restrict__ fc1w,
                                              const float* __restrict__ fc1b,
                                              const float* __restrict__ fc2w,
                                              const float* __restrict__ fc2b,
                                              float* __restrict__ out) {
  const int t   = blockIdx.x;
  const int tid = threadIdx.x;
  const float* v = hfin + ((size_t)(Bsz - 1) * Tsz + t) * TWOH;

  __shared__ float sv[TWOH];
  __shared__ float red[256];
  __shared__ float y[256];

  const float a0 = v[tid], a1 = v[tid + 256];

  red[tid] = a0 + a1;
  __syncthreads();
  for (int s = 128; s > 0; s >>= 1) {
    if (tid < s) red[tid] += red[tid + s];
    __syncthreads();
  }
  const float mu = red[0] * (1.0f / TWOH);
  __syncthreads();

  const float d0 = a0 - mu, d1 = a1 - mu;
  red[tid] = d0 * d0 + d1 * d1;
  __syncthreads();
  for (int s = 128; s > 0; s >>= 1) {
    if (tid < s) red[tid] += red[tid + s];
    __syncthreads();
  }
  const float rstd = rsqrtf(red[0] * (1.0f / TWOH) + 1e-5f);
  __syncthreads();

  sv[tid]       = d0 * rstd * lng[tid]       + lnb[tid];
  sv[tid + 256] = d1 * rstd * lng[tid + 256] + lnb[tid + 256];
  __syncthreads();

  float a = fc1b[tid];
  const float* wr = fc1w + (size_t)tid * TWOH;
  for (int k = 0; k < TWOH; ++k) a += sv[k] * wr[k];
  y[tid] = fmaxf(a, 0.0f);
  __syncthreads();

  if (tid < OUTsz) {
    float z = fc2b[tid];
    const float* w2 = fc2w + (size_t)tid * Hh;
    for (int k = 0; k < Hh; ++k) z += y[k] * w2[k];
    out[(size_t)t * OUTsz + tid] = sigf(z);
  }
}

// ---------------- host launcher ----------------
extern "C" void kernel_launch(void* const* d_in, const int* in_sizes, int n_in,
                              void* d_out, int out_size, void* d_ws, size_t ws_size,
                              hipStream_t stream) {
  const float* x     = (const float*)d_in[0];
  const float* w_ih0 = (const float*)d_in[1];
  const float* w_hh0 = (const float*)d_in[2];
  const float* b_ih0 = (const float*)d_in[3];
  const float* b_hh0 = (const float*)d_in[4];
  const float* w_ih  = (const float*)d_in[5];
  const float* w_hh  = (const float*)d_in[6];
  const float* b_ih  = (const float*)d_in[7];
  const float* b_hh  = (const float*)d_in[8];
  const float* lng   = (const float*)d_in[9];
  const float* lnb   = (const float*)d_in[10];
  const float* fc1w  = (const float*)d_in[11];
  const float* fc1b  = (const float*)d_in[12];
  const float* fc2w  = (const float*)d_in[13];
  const float* fc2b  = (const float*)d_in[14];

  char* ws = (char*)d_ws;
  size_t off = 0;
  float*  xg  = (float*)(ws + off);  off += (size_t)2 * MTOT * FOURH * 4;   // per-dir xg
  float*  hf  = (float*)(ws + off);  off += (size_t)MTOT * TWOH * 4;        // layer h (f32)
  __bf16* bfA = (__bf16*)(ws + off); off += (size_t)MTOT * TWOH * 2;        // layer h ping
  __bf16* bfB = (__bf16*)(ws + off); off += (size_t)MTOT * TWOH * 2;        // layer h pong
  __bf16* xbf = (__bf16*)(ws + off); off += (size_t)MTOT * IN_ * 2;         // x in bf16
  __bf16* wih = (__bf16*)(ws + off); off += ((size_t)2 * FOURH * IN_ + (size_t)8 * FOURH * TWOH) * 2;
  __bf16* whh = (__bf16*)(ws + off); off += (size_t)10 * FOURH * Hh * 2;
  float*  bsum = (float*)(ws + off); off += (size_t)10 * FOURH * 4;
  (void)ws_size; (void)in_sizes; (void)n_in; (void)out_size;

  // one-shot (per call) weight/input conversions
  k_f2bf<<<1024, 256, 0, stream>>>(x, xbf, MTOT * IN_);
  k_f2bf<<<1024, 256, 0, stream>>>(w_ih0, wih, 2 * FOURH * IN_);
  k_f2bf<<<4096, 256, 0, stream>>>(w_ih, wih + 2 * FOURH * IN_, 8 * FOURH * TWOH);
  k_f2bf<<<1024, 256, 0, stream>>>(w_hh0, whh, 2 * FOURH * Hh);
  k_f2bf<<<2048, 256, 0, stream>>>(w_hh, whh + 2 * FOURH * Hh, 8 * FOURH * Hh);
  k_biassum<<<8, 256, 0, stream>>>(b_ih0, b_hh0, bsum, 2 * FOURH);
  k_biassum<<<32, 256, 0, stream>>>(b_ih, b_hh, bsum + 2 * FOURH, 8 * FOURH);

  const __bf16* in_bf = xbf;
  int K = IN_;
  __bf16* outs[2] = {bfA, bfB};

  for (int l = 0; l < Lnum; ++l) {
    const __bf16* wih_l = (l == 0) ? wih
        : wih + 2 * FOURH * IN_ + (size_t)(l - 1) * 2 * FOURH * TWOH;
    const __bf16* whh_l = whh + (size_t)l * 2 * FOURH * Hh;
    const float*  bs_l  = bsum + l * 2 * FOURH;

    dim3 gproj(MTOT / 32, 4, 2);                 // 1024 x 4 x dirs
    k_proj<<<gproj, 128, 0, stream>>>(in_bf, K, wih_l, bs_l, xg);

    __bf16* ob = outs[l & 1];
    const int last = (l == Lnum - 1);
    k_scan<<<2, 1024, 0, stream>>>(xg, whh_l, hf, ob, last, !last);

    in_bf = ob;
    K = TWOH;
  }

  k_head<<<Tsz, 256, 0, stream>>>(hf, lng, lnb, fc1w, fc1b, fc2w, fc2b, (float*)d_out);
}
